// SlatewiseGRU_46634754900493
// MI455X (gfx1250) — matrix operand order repaired and checked by reading, not compiled, not run on hardware
//
#include <hip/hip_runtime.h>
#include <hip/hip_bf16.h>

// ---------------------------------------------------------------------------
// SlatewiseGRU on MI455X (gfx1250): bf16 WMMA recurrence, f32 accumulate.
//   N = 8192 slates, L = 32 steps, D = 128, gates = 3*128 = 384.
//   gate[m, j] = sum_k A[m,k] * W[j,k]   (A = x_t or h, W row-major [384,128])
// Per workgroup: 32 slates, 8 waves; wave w owns gate columns [16w, 16w+16).
// Accumulators start at inline 0 (no bias-splat movs); biases folded into the
// elementwise stage. Wave-local h columns kept in f32 registers across steps.
// ---------------------------------------------------------------------------

typedef __attribute__((ext_vector_type(16))) __bf16 v16bf;
typedef __attribute__((ext_vector_type(8)))  __bf16 v8bf;
typedef __attribute__((ext_vector_type(8)))  float  v8f;

#define DIM      128
#define LSTEPS   32
#define MWG      32          // slates per workgroup
#define NGROUPS  (MWG / 16)  // 16-row groups per wave
#define NTHREADS 256         // 8 waves
#define STRIDE   136         // padded LDS row stride (bf16 elems): 272 B = 68 banks

// LDS carve sizes (elements)
#define W_ELEMS   (384 * STRIDE)   // one weight matrix, bf16
#define T_ELEMS   (MWG * STRIDE)   // one 32x128 tile, bf16
static constexpr size_t SMEM_BYTES =
    (2 * W_ELEMS + 3 * T_ELEMS) * sizeof(__bf16) +   // Wih, Whh, X, H0, H1
    (384 + 384 + 128 + MWG) * sizeof(float);          // b_ih, b_hh, w_out, out_acc

__device__ __forceinline__ v16bf ld_frag(const __bf16* p, int off0, int off1) {
  // two 16-byte LDS loads -> one 16-element bf16 fragment (ds_load_b128 x2)
  v8bf lo = *(const v8bf*)(p + off0);
  v8bf hi = *(const v8bf*)(p + off1);
  return __builtin_shufflevector(lo, hi, 0,1,2,3,4,5,6,7,8,9,10,11,12,13,14,15);
}

__device__ __forceinline__ float sigmoid_(float x) { return 1.0f / (1.0f + __expf(-x)); }
__device__ __forceinline__ float tanh_(float x)    { return 2.0f / (1.0f + __expf(-2.0f * x)) - 1.0f; }

__global__ __launch_bounds__(NTHREADS, 1)
void slatewise_gru_wmma(const float* __restrict__ item_embs,   // [N, L, D]
                        const float* __restrict__ user_embs,   // [N, D]
                        const float* __restrict__ W_ih,        // [384, 128]
                        const float* __restrict__ W_hh,        // [384, 128]
                        const float* __restrict__ b_ih,        // [384]
                        const float* __restrict__ b_hh,        // [384]
                        const float* __restrict__ W_out,       // [128]
                        const float* __restrict__ b_out,       // [1]
                        float* __restrict__ out)               // [N, L]
{
  extern __shared__ char smem_raw[];
  __bf16* sWih = (__bf16*)smem_raw;
  __bf16* sWhh = sWih + W_ELEMS;
  __bf16* sX   = sWhh + W_ELEMS;
  __bf16* sHa  = sX   + T_ELEMS;
  __bf16* sHb  = sHa  + T_ELEMS;
  float*  sBih  = (float*)(sHb + T_ELEMS);
  float*  sBhh  = sBih + 384;
  float*  sWout = sBhh + 384;
  float*  sOut  = sWout + 128;

  const int tid  = threadIdx.x;
  const int lane = tid & 31;
  const int wave = tid >> 5;
  const int half = lane >> 4;       // 0: lanes 0-15, 1: lanes 16-31
  const int ln   = lane & 15;
  const int colbase = wave * 16;    // this wave's gate-column slice
  const int n0 = blockIdx.x * MWG;  // first slate of this workgroup

  // ---- one-time: weights -> bf16 LDS (padded rows), biases, W_out ----------
  for (int fi = tid; fi < 384 * (DIM / 4); fi += NTHREADS) {   // 12288 float4
    const int row = fi >> 5, c4 = fi & 31;
    float4 a = ((const float4*)(W_ih + (size_t)row * DIM))[c4];
    __bf16* d = sWih + row * STRIDE + c4 * 4;
    d[0] = (__bf16)a.x; d[1] = (__bf16)a.y; d[2] = (__bf16)a.z; d[3] = (__bf16)a.w;
    float4 b = ((const float4*)(W_hh + (size_t)row * DIM))[c4];
    __bf16* e = sWhh + row * STRIDE + c4 * 4;
    e[0] = (__bf16)b.x; e[1] = (__bf16)b.y; e[2] = (__bf16)b.z; e[3] = (__bf16)b.w;
  }
  for (int i = tid; i < 384; i += NTHREADS) { sBih[i] = b_ih[i]; sBhh[i] = b_hh[i]; }
  if (tid < 128) sWout[tid] = W_out[tid];

  // ---- h0 = user embedding, bf16 ------------------------------------------
  for (int fi = tid; fi < MWG * (DIM / 4); fi += NTHREADS) {   // 1024 float4
    const int row = fi >> 5, c4 = fi & 31;
    float4 a = ((const float4*)(user_embs + (size_t)(n0 + row) * DIM))[c4];
    __bf16* d = sHa + row * STRIDE + c4 * 4;
    d[0] = (__bf16)a.x; d[1] = (__bf16)a.y; d[2] = (__bf16)a.z; d[3] = (__bf16)a.w;
  }
  __syncthreads();

  // ---- persistent f32 copy of this wave's h columns (C layout) ------------
  // VGPR v of group g <-> row g*16 + v + half*8, column colbase + ln.
  v8f hC[NGROUPS];
  #pragma unroll
  for (int g = 0; g < NGROUPS; ++g)
    #pragma unroll
    for (int v = 0; v < 8; ++v)
      hC[g][v] = user_embs[(size_t)(n0 + g * 16 + v + half * 8) * DIM + colbase + ln];

  __bf16* hcur = sHa;
  __bf16* hnxt = sHb;
  const float bias_out = b_out[0];

  // per-wave bias values for its column slice (uniform across VGPRs)
  const float brz_r = sBih[colbase + ln]       + sBhh[colbase + ln];
  const float brz_z = sBih[128 + colbase + ln] + sBhh[128 + colbase + ln];
  const float b_xn  = sBih[256 + colbase + ln];
  const float b_hn  = sBhh[256 + colbase + ln];
  const float wo    = sWout[colbase + ln];

  #pragma unroll 1
  for (int t = 0; t < LSTEPS; ++t) {
    // ---- stage x_t (f32 -> bf16) and zero output accumulator --------------
    if (tid < MWG) sOut[tid] = 0.0f;
    for (int fi = tid; fi < MWG * (DIM / 4); fi += NTHREADS) {
      const int row = fi >> 5, c4 = fi & 31;
      const float* src = item_embs + ((size_t)(n0 + row) * LSTEPS + t) * DIM;
      float4 a = ((const float4*)src)[c4];
      __bf16* d = sX + row * STRIDE + c4 * 4;
      d[0] = (__bf16)a.x; d[1] = (__bf16)a.y; d[2] = (__bf16)a.z; d[3] = (__bf16)a.w;
      if (t + 1 < LSTEPS)  // pull next step into cache (global_prefetch_b8)
        __builtin_prefetch(src + DIM + c4 * 4, 0, 0);
    }
    __syncthreads();

    // streams 4 B-fragments of weight rows [rowbase, rowbase+16) and MACs
    auto mac = [&](const __bf16* W, int rowbase, const v16bf* A, v8f acc) -> v8f {
      #pragma unroll
      for (int c = 0; c < 4; ++c) {
        const int off0 = (rowbase + ln) * STRIDE + c * 32 + half * 8;
        v16bf bfrag = ld_frag(W, off0, off0 + 16);
        acc = __builtin_amdgcn_wmma_f32_16x16x32_bf16(
                  false, A[c], false, bfrag, (short)0, acc, false, false);
      }
      return acc;
    };

    #pragma unroll
    for (int g = 0; g < NGROUPS; ++g) {        // 16-slate row groups
      // A fragments (registers, reused across all gate tiles of this wave)
      v16bf ax[4], ah[4];
      #pragma unroll
      for (int c = 0; c < 4; ++c) {
        const int off0 = (g * 16 + ln) * STRIDE + c * 32 + half * 8;
        ax[c] = ld_frag(sX,   off0, off0 + 16);
        ah[c] = ld_frag(hcur, off0, off0 + 16);
      }

      const int gr = colbase, gz = 128 + colbase, gn = 256 + colbase;

      // inline-0 accumulators: v_wmma ..., 0 (no bias-splat movs)
      v8f ar = {}, az = {}, xn = {}, hn = {};
      ar = mac(sWih, gr, ax, ar);
      ar = mac(sWhh, gr, ah, ar);
      az = mac(sWih, gz, ax, az);
      az = mac(sWhh, gz, ah, az);
      xn = mac(sWih, gn, ax, xn);
      hn = mac(sWhh, gn, ah, hn);

      // ---- elementwise GRU update (biases folded in here) -----------------
      #pragma unroll
      for (int v = 0; v < 8; ++v) {
        const int row = g * 16 + v + half * 8;
        const float r  = sigmoid_(ar[v] + brz_r);
        const float z  = sigmoid_(az[v] + brz_z);
        const float nv = tanh_((xn[v] + b_xn) + r * (hn[v] + b_hn));
        const float hv = (1.0f - z) * nv + z * hC[g][v];
        hC[g][v] = hv;
        hnxt[row * STRIDE + colbase + ln] = (__bf16)hv;
        float s = hv * wo;                      // 16-lane row reduction
        s += __shfl_xor(s, 1, 32);
        s += __shfl_xor(s, 2, 32);
        s += __shfl_xor(s, 4, 32);
        s += __shfl_xor(s, 8, 32);
        if (ln == 0) atomicAdd(&sOut[row], s);  // ds_add_f32
      }
    }
    __syncthreads();

    if (tid < MWG) out[(size_t)(n0 + tid) * LSTEPS + t] = sOut[tid] + bias_out;

    __bf16* tmp = hcur; hcur = hnxt; hnxt = tmp;
  }
}

extern "C" void kernel_launch(void* const* d_in, const int* in_sizes, int n_in,
                              void* d_out, int out_size, void* d_ws, size_t ws_size,
                              hipStream_t stream) {
  (void)in_sizes; (void)n_in; (void)out_size; (void)d_ws; (void)ws_size;
  const float* item_embs = (const float*)d_in[0];
  const float* user_embs = (const float*)d_in[1];
  const float* W_ih      = (const float*)d_in[2];
  const float* W_hh      = (const float*)d_in[3];
  const float* b_ih      = (const float*)d_in[4];
  const float* b_hh      = (const float*)d_in[5];
  const float* W_out     = (const float*)d_in[6];
  const float* b_out     = (const float*)d_in[7];
  float* out = (float*)d_out;

  // 8192 slates / 32 per WG = 256 workgroups; ~239 KB dynamic LDS each.
  hipFuncSetAttribute((const void*)slatewise_gru_wmma,
                      hipFuncAttributeMaxDynamicSharedMemorySize,
                      (int)SMEM_BYTES);
  slatewise_gru_wmma<<<dim3(8192 / MWG), dim3(NTHREADS), SMEM_BYTES, stream>>>(
      item_embs, user_embs, W_ih, W_hh, b_ih, b_hh, W_out, b_out, out);
}
